// FlexAttentionWithRoPE_31482110279735
// MI455X (gfx1250) — matrix operand order repaired and checked by reading, not compiled, not run on hardware
//
#include <hip/hip_runtime.h>
#include <math.h>

typedef __attribute__((ext_vector_type(16))) _Float16     v16h;
typedef __attribute__((ext_vector_type(8)))  float        v8f;
typedef __attribute__((ext_vector_type(4)))  float        f32x4;
typedef __attribute__((ext_vector_type(4)))  unsigned int u32x4;
typedef __attribute__((ext_vector_type(8)))  int          i32x8;
typedef __attribute__((ext_vector_type(4)))  int          i32x4;

#define B_  2
#define H_  16
#define L_  2048
#define D_  64
#define F_  32
#define KT  64    // keys per tile
#define QT  64    // q rows per workgroup (16 per wave)
#define NW  4     // waves per workgroup

#if defined(__has_builtin)
#if __has_builtin(__builtin_amdgcn_tensor_load_to_lds)
#define HAVE_TDM 1
#endif
#endif
#ifndef HAVE_TDM
#define HAVE_TDM 0
#endif

// Flat LDS addresses carry the LDS byte offset in their low 32 bits
// (aperture lives in bits 63:32), so truncation yields the D# lds_addr.
#define LDS_OFF(p) ((unsigned)(unsigned long long)(const void*)(p))

// D = A*B + C, f16 inputs, f32 accum
static __device__ __forceinline__ v8f wmma_f16(v16h a, v16h b, v8f c) {
    return __builtin_amdgcn_wmma_f32_16x16x32_f16(false, a, false, b, (short)0, c,
                                                  false, false);
}

// DPP16 butterfly: xor1=quad_perm(1,0,3,2)=0xB1, xor2=quad_perm(2,3,0,1)=0x4E,
// xor7=row_half_mirror=0x141, xor15=row_mirror=0x140. Subgroup growth
// 2->4->8->16: full 16-lane row reduction confined to each wave32 half.
template <int CTRL>
static __device__ __forceinline__ float dpp_swap(float x) {
    return __builtin_bit_cast(float, __builtin_amdgcn_update_dpp(
        0, __builtin_bit_cast(int, x), CTRL, 0xF, 0xF, true));
}
static __device__ __forceinline__ float red16_max(float x) {
    x = fmaxf(x, dpp_swap<0xB1>(x));
    x = fmaxf(x, dpp_swap<0x4E>(x));
    x = fmaxf(x, dpp_swap<0x141>(x));
    x = fmaxf(x, dpp_swap<0x140>(x));
    return x;
}
static __device__ __forceinline__ float red16_add(float x) {
    x = x + dpp_swap<0xB1>(x);
    x = x + dpp_swap<0x4E>(x);
    x = x + dpp_swap<0x141>(x);
    x = x + dpp_swap<0x140>(x);
    return x;
}

static __device__ __forceinline__ void wait_tensorcnt0() {
#if __has_builtin(__builtin_amdgcn_s_wait_tensorcnt)
    __builtin_amdgcn_s_wait_tensorcnt(0);
#else
    asm volatile("s_wait_tensorcnt 0" ::: "memory");
#endif
}

// Issue one TDM tile load. tile0_8B = tile_dim0 in 8-byte units,
// tile1 = tile_dim1 rows (0 => 1-D), stride0_8B = tensor_dim0_stride.
// Tensor dims set huge so no OOB clamping applies (tiles are always in range).
static __device__ __forceinline__ void tdm_load_tile(const void* gsrc, unsigned lds_dst,
                                                     int tile0_8B, int tile1,
                                                     int stride0_8B) {
#if HAVE_TDM
    unsigned long long ga = (unsigned long long)gsrc;
    u32x4 g0;
    g0[0] = 1u;                                     // count=1, user descriptor
    g0[1] = lds_dst;                                // lds_addr
    g0[2] = (unsigned)ga;                           // global_addr[31:0]
    g0[3] = (unsigned)((ga >> 32) & 0x01FFFFFFu)    // global_addr[56:32]
          | (2u << 30);                             // type = 2
    const int td = 1 << 20;                         // huge tensor_dim0/1
    i32x8 g1;
    g1[0] = 3 << 16;                                // wg_mask=0, data_size=8B
    g1[1] = (td & 0xFFFF) << 16;                    // tensor_dim0[15:0]
    g1[2] = ((td >> 16) & 0xFFFF) | ((td & 0xFFFF) << 16);  // dim0 hi | dim1 lo
    g1[3] = ((td >> 16) & 0xFFFF) | (tile0_8B << 16);       // dim1 hi | tile_dim0
    g1[4] = tile1 & 0xFFFF;                         // tile_dim1 (tile_dim2 = 0)
    g1[5] = stride0_8B;                             // tensor_dim0_stride[31:0]
    g1[6] = 0;
    g1[7] = 0;
    i32x4 z4 = (i32x4)0;
#if __clang_major__ >= 23
    i32x8 z8 = (i32x8)0;
    __builtin_amdgcn_tensor_load_to_lds(g0, g1, z4, z4, z8, 0);
#else
    __builtin_amdgcn_tensor_load_to_lds(g0, g1, z4, z4, 0);
#endif
#else
    (void)gsrc; (void)lds_dst; (void)tile0_8B; (void)tile1; (void)stride0_8B;
#endif
}

// ---------------------------------------------------------------------------
// Prep kernel (runs once): RoPE K -> f16 [b,h,key,d]; V -> f16 [b,h,d,key].
// ---------------------------------------------------------------------------
__global__ __launch_bounds__(128)
void rope_prep_kernel(const float* __restrict__ k,
                      const float* __restrict__ v,
                      const float* __restrict__ pos,
                      const float* __restrict__ freqs,
                      _Float16* __restrict__ krot,
                      _Float16* __restrict__ vtg)
{
    const int tid = threadIdx.x;
    const int h = blockIdx.y, b = blockIdx.z;
    const int row   = blockIdx.x * 64 + (tid >> 1);   // key row
    const int fbase = (tid & 1) << 4;                 // f in [fbase, fbase+16)
    const size_t bh = (size_t)(b * H_ + h) * L_;

    const float p0 = pos[(b * L_ + row) * 2 + 0];
    const float p1 = pos[(b * L_ + row) * 2 + 1];
    const float* kp = k + (bh + row) * D_;
    _Float16*    ko = krot + (bh + row) * D_;
    f32x4 x1v[4], x2v[4];
    #pragma unroll
    for (int t = 0; t < 4; ++t) {
        x1v[t] = *(const f32x4*)(kp + fbase + 4 * t);
        x2v[t] = *(const f32x4*)(kp + fbase + 32 + 4 * t);
    }
    #pragma unroll
    for (int i = 0; i < 16; ++i) {
        int f = fbase + i;
        float th = p0 * freqs[(h * F_ + f) * 2 + 0]
                 + p1 * freqs[(h * F_ + f) * 2 + 1];
        float s, c; __sincosf(th, &s, &c);
        float x1 = x1v[i >> 2][i & 3];
        float x2 = x2v[i >> 2][i & 3];
        ko[f]      = (_Float16)(x1 * c - x2 * s);
        ko[f + 32] = (_Float16)(x2 * c + x1 * s);
    }

    // V: convert + transpose into [d][key]
    const int vcb = (tid & 1) << 5;                   // cols [vcb, vcb+32)
    const float* vp = v + (bh + row) * D_;
    _Float16*    vo = vtg + (size_t)(b * H_ + h) * D_ * L_;
    #pragma unroll
    for (int i = 0; i < 32; i += 4) {
        f32x4 vv = *(const f32x4*)(vp + vcb + i);
        vo[(size_t)(vcb + i + 0) * L_ + row] = (_Float16)vv[0];
        vo[(size_t)(vcb + i + 1) * L_ + row] = (_Float16)vv[1];
        vo[(size_t)(vcb + i + 2) * L_ + row] = (_Float16)vv[2];
        vo[(size_t)(vcb + i + 3) * L_ + row] = (_Float16)vv[3];
    }
}

// ---------------------------------------------------------------------------
// Flash-attention kernel: f16 tiles DMA'd (TDM) into double-buffered LDS.
// ---------------------------------------------------------------------------
__global__ __launch_bounds__(128)
void fa_rope_kernel(const float* __restrict__ q,
                    const float* __restrict__ pos,
                    const float* __restrict__ freqs,
                    const _Float16* __restrict__ krot,
                    const _Float16* __restrict__ vtg,
                    float* __restrict__ out)
{
    __shared__ _Float16 ktb[2][KT][D_];    // RoPE'd K tiles [key][d]
    __shared__ _Float16 vtb[2][D_][KT];    // V tiles [col][key]
    __shared__ _Float16 pbuf[NW][16][KT];  // per-wave P transpose scratch

    const int tid   = threadIdx.x;
    const int lane  = tid & 31;
    const int wave  = tid >> 5;
    const int qtile = blockIdx.x;
    const int h     = blockIdx.y;
    const int b     = blockIdx.z;

    const int ln16 = lane & 15;
    const int hiA  = (lane >> 4) << 3;   // lane-half row offset (0 or 8)
    const int hiB  = (lane >> 4) << 4;   // B-layout K offset (0 or 16)
    const int row8 = hiA;                // C/D row offset for this lane half

    const size_t bh  = (size_t)(b * H_ + h) * L_;
    const size_t bhd = (size_t)(b * H_ + h) * D_;

    // ---- load + RoPE + pre-scale this wave's 16x64 Q tile into A layout ----
    // scale folds 1/sqrt(D) and log2(e): softmax runs in base-2 domain.
    const int   qrow  = qtile * QT + wave * 16 + ln16;
    const float scale = 0.125f * 1.44269504088896341f;
    const float pq0 = pos[(b * L_ + qrow) * 2 + 0];
    const float pq1 = pos[(b * L_ + qrow) * 2 + 1];
    const float* qp = q + (bh + qrow) * D_;
    v16h qa0, qa1;
    #pragma unroll
    for (int e = 0; e < 16; ++e) {
        int f = e + hiB;                                   // 0..31
        float th = pq0 * freqs[(h * F_ + f) * 2 + 0]
                 + pq1 * freqs[(h * F_ + f) * 2 + 1];
        float s, c; __sincosf(th, &s, &c);
        float x1 = qp[f] * scale, x2 = qp[f + 32] * scale;
        qa0[e] = (_Float16)(x1 * c - x2 * s);              // d = f
        qa1[e] = (_Float16)(x2 * c + x1 * s);              // d = f + 32
    }

    // ---- flash-attention state ----
    v8f Oc[4]; float m_r[8], l_r[8];
    #pragma unroll
    for (int c = 0; c < 4; ++c) Oc[c] = (v8f){};
    #pragma unroll
    for (int r = 0; r < 8; ++r) { m_r[r] = -INFINITY; l_r[r] = 0.f; }

    const int ntiles = L_ / KT;          // 32

#if !HAVE_TDM
    uint4 kreg[4], vreg[4];
    auto gload = [&](int jj) {
        const uint4* ks = (const uint4*)(krot + (bh + (size_t)jj * KT) * D_);
        #pragma unroll
        for (int t = 0; t < 4; ++t) kreg[t] = ks[tid + 128 * t];
        const uint4* vs = (const uint4*)(vtg + (bhd + (tid >> 1)) * L_
                                         + (size_t)jj * KT + (tid & 1) * 32);
        #pragma unroll
        for (int t = 0; t < 4; ++t) vreg[t] = vs[t];
    };
    auto lstore = [&](int buf) {
        uint4* kd = (uint4*)&ktb[buf][0][0];
        #pragma unroll
        for (int t = 0; t < 4; ++t) kd[tid + 128 * t] = kreg[t];
        uint4* vd = (uint4*)&vtb[buf][tid >> 1][(tid & 1) * 32];
        #pragma unroll
        for (int t = 0; t < 4; ++t) vd[t] = vreg[t];
    };
    gload(0);
#else
    if (wave == 0) {   // DMA tile 0: K 8KB contiguous (1-D), V 64x128B (2-D)
        tdm_load_tile(krot + bh * D_, LDS_OFF(&ktb[0][0][0]), 1024, 0, 1024);
        tdm_load_tile(vtg + bhd * L_, LDS_OFF(&vtb[0][0][0]), 16, 64, L_ * 2 / 8);
    }
#endif

    for (int j = 0; j < ntiles; ++j) {
        const int buf = j & 1;
#if HAVE_TDM
        if (wave == 0) wait_tensorcnt0();     // tile j landed in LDS
        __syncthreads();
        if (j + 1 < ntiles && wave == 0) {    // DMA tile j+1 under compute
            tdm_load_tile(krot + (bh + (size_t)(j + 1) * KT) * D_,
                          LDS_OFF(&ktb[buf ^ 1][0][0]), 1024, 0, 1024);
            tdm_load_tile(vtg + bhd * L_ + (size_t)(j + 1) * KT,
                          LDS_OFF(&vtb[buf ^ 1][0][0]), 16, 64, L_ * 2 / 8);
        }
#else
        lstore(buf);
        __syncthreads();
        if (j + 1 < ntiles) gload(j + 1);     // global latency under compute
#endif

        // ---- S(16x64) = Qrot x Krot^T : 8 WMMAs ----
        v8f S[4];
        #pragma unroll
        for (int g = 0; g < 4; ++g) {
            v16h kb0 = *(const v16h*)&ktb[buf][g * 16 + ln16][hiB];
            v16h kb1 = *(const v16h*)&ktb[buf][g * 16 + ln16][32 + hiB];
            v8f s = (v8f){};
            s = wmma_f16(qa0, kb0, s);
            s = wmma_f16(qa1, kb1, s);
            S[g] = s;
        }

        // ---- online softmax (base-2 domain; DPP16 row reductions) ----
        float mt[8];
        #pragma unroll
        for (int r = 0; r < 8; ++r)
            mt[r] = red16_max(fmaxf(fmaxf(S[0][r], S[1][r]),
                                    fmaxf(S[2][r], S[3][r])));

        float rs[8];
        #pragma unroll
        for (int r = 0; r < 8; ++r) {
            float mn    = fmaxf(m_r[r], mt[r]);
            float alpha = exp2f(m_r[r] - mn);
            m_r[r] = mn;
            float acc = 0.f;
            #pragma unroll
            for (int g = 0; g < 4; ++g) {      // S becomes P in-place
                S[g][r] = exp2f(S[g][r] - mn);
                acc += S[g][r];
            }
            rs[r]   = acc;
            l_r[r] *= alpha;
            Oc[0][r] *= alpha; Oc[1][r] *= alpha;
            Oc[2][r] *= alpha; Oc[3][r] *= alpha;
        }
        #pragma unroll
        for (int r = 0; r < 8; ++r) l_r[r] += red16_add(rs[r]);

        // ---- transpose P (D layout -> A layout) through per-wave LDS ----
        _Float16 (*pb)[KT] = pbuf[wave];
        #pragma unroll
        for (int r = 0; r < 8; ++r) {
            #pragma unroll
            for (int g = 0; g < 4; ++g)
                pb[r + row8][g * 16 + ln16] = (_Float16)S[g][r];
        }
        asm volatile("s_wait_dscnt 0" ::: "memory");  // intra-wave LDS RAW
        v16h pa0, pa1;
        #pragma unroll
        for (int e = 0; e < 16; ++e) {
            int kk = e + hiB;                         // A-layout K index
            pa0[e] = pb[ln16][kk];                    // keys 0..31
            pa1[e] = pb[ln16][32 + kk];               // keys 32..63
        }

        // ---- O(16x64) += P x V : 8 WMMAs ----
        #pragma unroll
        for (int c = 0; c < 4; ++c) {
            v16h vb0 = *(const v16h*)&vtb[buf][c * 16 + ln16][hiB];
            v16h vb1 = *(const v16h*)&vtb[buf][c * 16 + ln16][32 + hiB];
            Oc[c] = wmma_f16(pa0, vb0, Oc[c]);
            Oc[c] = wmma_f16(pa1, vb1, Oc[c]);
        }
    }

    // ---- normalize and write out (C/D layout: row = r+row8, col = c*16+ln16) --
    float* op = out + bh * D_;
    #pragma unroll
    for (int r = 0; r < 8; ++r) {
        int   row = qtile * QT + wave * 16 + r + row8;
        float inv = 1.0f / l_r[r];
        #pragma unroll
        for (int c = 0; c < 4; ++c)
            op[(size_t)row * D_ + c * 16 + ln16] = Oc[c][r] * inv;
    }
}

extern "C" void kernel_launch(void* const* d_in, const int* in_sizes, int n_in,
                              void* d_out, int out_size, void* d_ws, size_t ws_size,
                              hipStream_t stream) {
    const float* q     = (const float*)d_in[0];
    const float* k     = (const float*)d_in[1];
    const float* v     = (const float*)d_in[2];
    const float* pos   = (const float*)d_in[3];
    const float* freqs = (const float*)d_in[4];
    float* out = (float*)d_out;
    (void)in_sizes; (void)n_in; (void)out_size; (void)ws_size;

    const size_t nkv = (size_t)B_ * H_ * L_ * D_;   // 4M elements, 8MB each
    _Float16* krot = (_Float16*)d_ws;
    _Float16* vtg  = krot + nkv;

    dim3 grid(L_ / QT, H_, B_);   // (32, 16, 2)
    dim3 block(128);              // 4 waves of 32
    rope_prep_kernel<<<grid, block, 0, stream>>>(k, v, pos, freqs, krot, vtg);
    fa_rope_kernel<<<grid, block, 0, stream>>>(q, pos, freqs, krot, vtg, out);
}
